// LinearChainDecoder_13383118094359
// MI455X (gfx1250) — matrix operand order repaired and checked by reading, not compiled, not run on hardware
//
#include <hip/hip_runtime.h>

typedef __attribute__((ext_vector_type(16))) _Float16 v16h;
typedef __attribute__((ext_vector_type(8)))  _Float16 v8h;
typedef __attribute__((ext_vector_type(4)))  _Float16 v4h;
typedef __attribute__((ext_vector_type(8)))  float    v8f;

#define T_STEPS 1024
#define BATCH   1024
#define NS      64
#define TILE_B  16
#define PF      4                 // emission tiles in flight (async ring)
#define RP      72                // vbuf row pitch in halfs (bank-conflict-free)
#define TILE_FLOATS (TILE_B * NS)
#define TILE_BYTES  (TILE_FLOATS * 4)
#define LN2F 0.6931471805599453f

// Async-copy one 16x64 f32 emission tile (4KB contiguous) into the LDS ring.
// 256 x B128 chunks, 32 lanes -> 8 async instructions per wave per tile.
__device__ __forceinline__ void prefetch_tile(const float* __restrict__ em,
                                              unsigned lds_base, int t, int b0, int l) {
  const unsigned gbase = ((unsigned)t * (unsigned)(BATCH * NS) + (unsigned)b0 * NS) * 4u;
  const unsigned lbase = lds_base + (unsigned)(t & (PF - 1)) * TILE_BYTES;
#pragma unroll
  for (int it = 0; it < 8; ++it) {
    const unsigned off  = (unsigned)((it * 32 + l) * 16);
    const unsigned ldsa = lbase + off;
    const unsigned ga   = gbase + off;
    asm volatile("global_load_async_to_lds_b128 %0, %1, %2"
                 :: "v"(ldsa), "v"(ga), "s"(em) : "memory");
  }
}

__global__ __launch_bounds__(32)
void crf_forward_kernel(const float* __restrict__ em,   // [T,B,S]
                        const int*   __restrict__ lens, // [B]
                        const float* __restrict__ start,// [S]
                        const float* __restrict__ stop, // [S]
                        const float* __restrict__ trans,// [S,S]  trans[s][p]
                        float*       __restrict__ out)  // [B]
{
  __shared__ _Float16 vbuf[TILE_B * RP];        // lin = renormalized exp(alpha), A source
  __shared__ float    emb[PF * TILE_FLOATS];    // emission tile ring buffer

  const int l  = threadIdx.x;       // 0..31
  const int ln = l & 15;
  const int hi = l >> 4;            // 0: rows 0-7 / K 0-7,16-23 per 32-chunk, 1: rows 8-15 / K 8-15,24-31
  const int b0 = blockIdx.x * TILE_B;

  // ---- constant B fragments. K is stored permuted: k' = (k&15)*4 + (k>>4), so each
  // lane's 4 per-row lin values (s = nt*16+ln) are contiguous at k' = ln*4+nt.
  // B[k'][n] = exp(trans[n][k]); identical permutation on A and B leaves the sum invariant.
  v16h Bf[4][2];
#pragma unroll
  for (int nt = 0; nt < 4; ++nt) {
    const int n = nt * 16 + ln;
#pragma unroll
    for (int kk = 0; kk < 2; ++kk) {
      const int kb = kk * 32 + hi * 8;
#pragma unroll
      for (int j = 0; j < 16; ++j) {
        const int kp = kb + (j & 7) + ((j >> 3) << 4);   // permuted K index
        const int k  = (kp & 3) * 16 + (kp >> 2);
        Bf[nt][kk][j] = (_Float16)__expf(trans[n * NS + k]);
      }
    }
  }

  int lenr[8];
#pragma unroll
  for (int r = 0; r < 8; ++r) lenr[r] = lens[b0 + hi * 8 + r];

  const unsigned emb_lds = (unsigned)(unsigned long long)(&emb[0]);

  // Prologue: fill the async ring.
#pragma unroll
  for (int t = 0; t < PF; ++t) prefetch_tile(em, emb_lds, t, b0, l);

  // ---- t = 0: alpha0 = start + em0; switch to linear space: Cr = rowmax, lin = exp(a0-Cr)
  asm volatile("s_wait_asynccnt 24" ::: "memory");   // oldest tile (8*(PF-1)) complete

  float Cr[8], Cfin[8];         // running log-offset per row; snapshot at t == len-1
  float afin[4][8];             // snapshot of lin at t == len-1
  {
    float a0[4][8];
#pragma unroll
    for (int nt = 0; nt < 4; ++nt) {
      const int s = nt * 16 + ln;
      const float sv = start[s];
#pragma unroll
      for (int r = 0; r < 8; ++r)
        a0[nt][r] = sv + emb[(hi * 8 + r) * NS + s];
    }
#pragma unroll
    for (int r = 0; r < 8; ++r) {
      float mx = fmaxf(fmaxf(a0[0][r], a0[1][r]), fmaxf(a0[2][r], a0[3][r]));
      mx = fmaxf(mx, __shfl_xor(mx, 1));
      mx = fmaxf(mx, __shfl_xor(mx, 2));
      mx = fmaxf(mx, __shfl_xor(mx, 4));
      mx = fmaxf(mx, __shfl_xor(mx, 8));
      Cr[r] = mx; Cfin[r] = mx;
      v4h pk;
#pragma unroll
      for (int nt = 0; nt < 4; ++nt) {
        const float lin = __expf(a0[nt][r] - mx);
        afin[nt][r] = lin;
        pk[nt] = (_Float16)lin;
      }
      *(v4h*)&vbuf[(hi * 8 + r) * RP + ln * 4] = pk;   // permuted-K layout
    }
  }

  // Preload A fragments for step 1 (software pipeline over the LDS latency).
  v8h pa0 = *(const v8h*)&vbuf[ln * RP + hi * 8];
  v8h pa1 = *(const v8h*)&vbuf[ln * RP + hi * 8 + 16];
  v8h pa2 = *(const v8h*)&vbuf[ln * RP + 32 + hi * 8];
  v8h pa3 = *(const v8h*)&vbuf[ln * RP + 32 + hi * 8 + 16];

  prefetch_tile(em, emb_lds, PF, b0, l);             // refill slot 0

  // ---- main scan (branchless body; dead rows keep evolving, snapshot captures them) ----
#pragma unroll 1
  for (int t = 1; t < T_STEPS; ++t) {
    asm volatile("s_wait_asynccnt 24" ::: "memory"); // emission tile t ready
    const float* ebuf = &emb[(t & (PF - 1)) * TILE_FLOATS];

    // Phase A: acc = lin @ exp(trans)^T   (f16 WMMA, f32 accum, permuted K)
    const v16h a0 = __builtin_shufflevector(pa0, pa1,
                      0,1,2,3,4,5,6,7,8,9,10,11,12,13,14,15);
    const v16h a1 = __builtin_shufflevector(pa2, pa3,
                      0,1,2,3,4,5,6,7,8,9,10,11,12,13,14,15);
    v8f acc[4] = {};
#pragma unroll
    for (int nt = 0; nt < 4; ++nt)
      acc[nt] = __builtin_amdgcn_wmma_f32_16x16x32_f16(
          false, a0, false, Bf[nt][0], (short)0, acc[nt], false, false);
#pragma unroll
    for (int nt = 0; nt < 4; ++nt)
      acc[nt] = __builtin_amdgcn_wmma_f32_16x16x32_f16(
          false, a1, false, Bf[nt][1], (short)0, acc[nt], false, false);

    // Phase B: q = acc * exp(em)  (exps depend only on prefetched em -> overlap WMMA)
    float q[4][8];
#pragma unroll
    for (int nt = 0; nt < 4; ++nt)
#pragma unroll
      for (int r = 0; r < 8; ++r)
        q[nt][r] = acc[nt][r] * __expf(ebuf[(hi * 8 + r) * NS + nt * 16 + ln]);

    // Phase C: power-of-two renorm from row-max exponent; no log/rcp on the scan path.
#pragma unroll
    for (int r = 0; r < 8; ++r) {
      float mx = fmaxf(fmaxf(q[0][r], q[1][r]), fmaxf(q[2][r], q[3][r]));
      mx = fmaxf(mx, __shfl_xor(mx, 1));
      mx = fmaxf(mx, __shfl_xor(mx, 2));
      mx = fmaxf(mx, __shfl_xor(mx, 4));
      mx = fmaxf(mx, __shfl_xor(mx, 8));
      const unsigned eb = (__float_as_uint(mx) >> 23) & 0xffu;     // biased exponent
      Cr[r] += (float)((int)eb - 126) * LN2F;                       // C += e*ln2
      const float scale = __uint_as_float((253u - eb) << 23);       // 2^-e
      const bool cap = (t == lenr[r] - 1);                          // snapshot point
      Cfin[r] = cap ? Cr[r] : Cfin[r];
      v4h pk;
#pragma unroll
      for (int nt = 0; nt < 4; ++nt) {
        const float lin = q[nt][r] * scale;                         // in (0,1]
        afin[nt][r] = cap ? lin : afin[nt][r];
        pk[nt] = (_Float16)lin;
      }
      *(v4h*)&vbuf[(hi * 8 + r) * RP + ln * 4] = pk;
    }

    // Preload next step's A fragments (wave-private LDS is in-order: loads see stores).
    pa0 = *(const v8h*)&vbuf[ln * RP + hi * 8];
    pa1 = *(const v8h*)&vbuf[ln * RP + hi * 8 + 16];
    pa2 = *(const v8h*)&vbuf[ln * RP + 32 + hi * 8];
    pa3 = *(const v8h*)&vbuf[ln * RP + 32 + hi * 8 + 16];

    if (t + PF < T_STEPS) prefetch_tile(em, emb_lds, t + PF, b0, l);
  }

  // ---- epilogue: out[b] = Cfin + log( sum_s afin[s] * exp(stop[s]) ) ----
  float estop[4];
#pragma unroll
  for (int nt = 0; nt < 4; ++nt) estop[nt] = __expf(stop[nt * 16 + ln]);

#pragma unroll
  for (int r = 0; r < 8; ++r) {
    float sm = 0.0f;
#pragma unroll
    for (int nt = 0; nt < 4; ++nt) sm += afin[nt][r] * estop[nt];
    sm += __shfl_xor(sm, 1);
    sm += __shfl_xor(sm, 2);
    sm += __shfl_xor(sm, 4);
    sm += __shfl_xor(sm, 8);
    if (ln == 0) out[b0 + hi * 8 + r] = Cfin[r] + __logf(sm);
  }
}

extern "C" void kernel_launch(void* const* d_in, const int* in_sizes, int n_in,
                              void* d_out, int out_size, void* d_ws, size_t ws_size,
                              hipStream_t stream) {
  const float* em    = (const float*)d_in[0]; // [T,B,S]
  const int*   lens  = (const int*)  d_in[1]; // [B]
  const float* start = (const float*)d_in[2]; // [S]
  const float* stop  = (const float*)d_in[3]; // [S]
  const float* trans = (const float*)d_in[4]; // [S,S]
  float* out = (float*)d_out;                 // [B]
  (void)in_sizes; (void)n_in; (void)out_size; (void)d_ws; (void)ws_size;

  crf_forward_kernel<<<BATCH / TILE_B, 32, 0, stream>>>(em, lens, start, stop, trans, out);
}
